// SAModule_21483426414943
// MI455X (gfx1250) — matrix operand order repaired and checked by reading, not compiled, not run on hardware
//
#include <hip/hip_runtime.h>
#include <hip/hip_bf16.h>
#include <math.h>

// Problem sizes (match reference)
#define Bc   8
#define Nn   4096
#define Ff   64
#define Ss   2048          // N * RATIO
#define Kk   64            // max neighbors
#define H1c  64
#define H2c  64
#define H3c  128
#define R2c  0.04f         // RADIUS^2
#define CAND_MAX 1024

// Packed f16 weight blob (in d_ws, produced by pack_weights_kernel).
// All matrices stored TRANSPOSED (N-major) so WMMA B-fragment loads are
// contiguous per lane (2x ds_load_b128 instead of 16x ds_load_u16):
//   [0,     6144)  W1t [64 n][96 k]  (k rows 67..95 zero padded)
//   [6144, 10240)  W2t [64 n][64 k]
//   [10240,18432)  W3t [128 n][64 k]
static constexpr int W1OFF = 0;
static constexpr int W2OFF = 64 * 96;
static constexpr int W3OFF = 64 * 96 + 64 * 64;
static constexpr int WTOT  = 64 * 96 + 64 * 64 + 128 * 64;   // 18432 halves = 36864 B

typedef __attribute__((ext_vector_type(16))) _Float16     v16h;
typedef __attribute__((ext_vector_type(8)))  float        v8f;
typedef __attribute__((ext_vector_type(4)))  unsigned int u32x4;
typedef __attribute__((ext_vector_type(8)))  int          i32x8;
typedef __attribute__((ext_vector_type(4)))  int          i32x4;

#define HAVE_TDM (__has_builtin(__builtin_amdgcn_tensor_load_to_lds) && \
                  __has_builtin(__builtin_amdgcn_s_wait_tensorcnt))

// ---------------------------------------------------------------------------
// WMMA fragment helpers (wave32, 16x16x32 f16; A/B use contiguous k = 16*half + s here)
// A (16x32, MxK): lane L -> half=L>>4, m=L&15; VGPR p holds K pair
//   kk(p) = 2p + 16*half
// B (32x16, KxN): lane L -> kh=L>>4, n=L&15; VGPR p holds K = 2p + 16*kh, +1
//   -> with W stored transposed (N-major), each lane reads 16 contiguous halves
// C/D (16x16 f32): VGPR r: lanes 0-15 -> M=r, lanes 16-31 -> M=r+8; N=lane&15
// ---------------------------------------------------------------------------
__device__ __forceinline__ v16h load_a_frag(const _Float16* base, int stride) {
  const int lane = threadIdx.x & 31;
  const int half = lane >> 4;
  const int m    = lane & 15;
  v16h a;
#pragma unroll
  for (int p = 0; p < 8; ++p) {
    const int kk = 2 * p + 16 * half;
    a[2 * p]     = base[m * stride + kk];
    a[2 * p + 1] = base[m * stride + kk + 1];
  }
  return a;
}

__device__ __forceinline__ v16h load_bt_frag(const _Float16* base, int kstride) {
  const int lane = threadIdx.x & 31;
  const int kh   = lane >> 4;
  const int n    = lane & 15;
  v16h v;
#pragma unroll
  for (int p = 0; p < 8; ++p) {
    const int kk = 2 * p + 16 * kh;
    v[2 * p]     = base[n * kstride + kk];
    v[2 * p + 1] = base[n * kstride + kk + 1];
  }
  return v;
}

__device__ __forceinline__ void mlp_tile(const _Float16* A, int astride,
                                         const _Float16* Bt, int bkstride,
                                         int kchunks, v8f& acc) {
#pragma unroll
  for (int kc = 0; kc < kchunks; ++kc) {
    const v16h a  = load_a_frag(A + kc * 32, astride);
    const v16h bb = load_bt_frag(Bt + kc * 32, bkstride);
    acc = __builtin_amdgcn_wmma_f32_16x16x32_f16(false, a, false, bb,
                                                 (short)0, acc, false, false);
  }
}

__device__ __forceinline__ void store_c_bias_relu_f16(_Float16* out, int stride,
                                                      v8f acc, const float* bias) {
  const int lane = threadIdx.x & 31;
  const int half = lane >> 4;
  const int n    = lane & 15;
#pragma unroll
  for (int r = 0; r < 8; ++r) {
    float v = acc[r] + bias[n];
    out[(r + 8 * half) * stride + n] = (_Float16)fmaxf(v, 0.0f);
  }
}

// ---------------------------------------------------------------------------
// Kernel 0: pack weights once into one transposed f16 blob in d_ws.
// ---------------------------------------------------------------------------
__global__ __launch_bounds__(256)
void pack_weights_kernel(const float* __restrict__ W1, const float* __restrict__ W2,
                         const float* __restrict__ W3, _Float16* __restrict__ wout) {
  const int i = blockIdx.x * 256 + threadIdx.x;
  if (i < W2OFF) {                       // W1t: [64 n][96 k], W1 is [67 k][64 n]
    const int n = i / 96, k = i % 96;
    wout[i] = (_Float16)((k < Ff + 3) ? W1[k * H1c + n] : 0.0f);
  } else if (i < W3OFF) {                // W2t: [64 n][64 k], W2 is [64 k][64 n]
    const int j = i - W2OFF;
    const int n = j >> 6, k = j & 63;
    wout[i] = (_Float16)W2[k * H2c + n];
  } else if (i < WTOT) {                 // W3t: [128 n][64 k], W3 is [64 k][128 n]
    const int j = i - W3OFF;
    const int n = j >> 6, k = j & 63;
    wout[i] = (_Float16)W3[k * H3c + n];
  }
}

// ---------------------------------------------------------------------------
// Kernel 1: farthest point sampling. One 1024-thread workgroup per cloud.
// ---------------------------------------------------------------------------
__global__ __launch_bounds__(1024)
void fps_kernel(const float* __restrict__ pos,
                float* __restrict__ posOut,
                float* __restrict__ batchOut) {
  const int b    = blockIdx.x;
  const int tid  = threadIdx.x;
  const int lane = tid & 31;
  const int wid  = tid >> 5;

  __shared__ float wv[32];
  __shared__ int   wi[32];
  __shared__ float sCur[3];

  float px[4], py[4], pz[4], dmin[4];
#pragma unroll
  for (int j = 0; j < 4; ++j) {
    const int p  = tid + j * 1024;
    const size_t o = ((size_t)b * Nn + p) * 3;
    px[j] = pos[o]; py[j] = pos[o + 1]; pz[j] = pos[o + 2];
    dmin[j] = INFINITY;
  }
  if (tid == 0) {
    const size_t o = (size_t)b * Nn * 3;
    sCur[0] = pos[o]; sCur[1] = pos[o + 1]; sCur[2] = pos[o + 2];
  }
  __syncthreads();

  for (int s = 0; s < Ss; ++s) {
    const float cx = sCur[0], cy = sCur[1], cz = sCur[2];
    if (tid == 0) {
      const int o = b * Ss + s;
      posOut[o * 3 + 0] = cx;
      posOut[o * 3 + 1] = cy;
      posOut[o * 3 + 2] = cz;
      batchOut[o] = (float)b;
    }
    float bv = -1.0f;
    int   bi = 0x7fffffff;
#pragma unroll
    for (int j = 0; j < 4; ++j) {
      const float dx = px[j] - cx, dy = py[j] - cy, dz = pz[j] - cz;
      const float d  = dx * dx + dy * dy + dz * dz;
      dmin[j] = fminf(dmin[j], d);
      if (dmin[j] > bv) { bv = dmin[j]; bi = tid + j * 1024; }
    }
#pragma unroll
    for (int o = 16; o > 0; o >>= 1) {
      const float ov = __shfl_xor(bv, o);
      const int   oi = __shfl_xor(bi, o);
      if (ov > bv || (ov == bv && oi < bi)) { bv = ov; bi = oi; }
    }
    if (lane == 0) { wv[wid] = bv; wi[wid] = bi; }
    __syncthreads();
    if (wid == 0) {
      bv = wv[lane]; bi = wi[lane];
#pragma unroll
      for (int o = 16; o > 0; o >>= 1) {
        const float ov = __shfl_xor(bv, o);
        const int   oi = __shfl_xor(bi, o);
        if (ov > bv || (ov == bv && oi < bi)) { bv = ov; bi = oi; }
      }
      if (lane == 0) {
        const size_t o = ((size_t)b * Nn + bi) * 3;
        sCur[0] = pos[o]; sCur[1] = pos[o + 1]; sCur[2] = pos[o + 2];
      }
    }
    __syncthreads();
  }
}

// ---------------------------------------------------------------------------
// Kernel 2: fused ball-query + nearest-K selection + PointNet MLP (WMMA) +
// masked max aggregation. One 256-thread (8 wave32) workgroup per centroid.
// Weights arrive in LDS via one TDM tensor_load_to_lds (36 KB f16 blob).
// ---------------------------------------------------------------------------
__global__ __launch_bounds__(256)
void sa_mlp_kernel(const float* __restrict__ x, const float* __restrict__ pos,
                   const _Float16* __restrict__ wall,
                   const float* __restrict__ b1, const float* __restrict__ b2,
                   const float* __restrict__ b3,
                   const float* __restrict__ posOut, float* __restrict__ xOut) {
  const int cid  = blockIdx.x;
  const int b    = cid / Ss;
  const int tid  = threadIdx.x;
  const int lane = tid & 31;
  const int wid  = tid >> 5;

  __shared__ _Float16 Wall[WTOT];     // f16 weight blob (TDM destination)
  __shared__ float    candD[CAND_MAX];
  __shared__ int      candI[CAND_MAX];
  __shared__ int      candCount;
  __shared__ int      nbrS[Kk];
  __shared__ int      sCnt;
  __shared__ _Float16 Hh[64 * 96];    // [K neighbors][F+3 padded to 96]
  __shared__ _Float16 O1h[64 * 64];
  __shared__ _Float16 O2h[64 * 64];
  __shared__ float    b1f[64], b2f[64], b3f[128];
  __shared__ float    pmax[4 * 128];  // per M-tile partial col max

  const _Float16* W1t = Wall + W1OFF; // [64 n][96 k]
  const _Float16* W2t = Wall + W2OFF; // [64 n][64 k]
  const _Float16* W3t = Wall + W3OFF; // [128 n][64 k]

  const float pcx = posOut[cid * 3 + 0];
  const float pcy = posOut[cid * 3 + 1];
  const float pcz = posOut[cid * 3 + 2];

  if (tid == 0) candCount = 0;
  __syncthreads();

  // ---- kick off TDM load of the 36KB f16 weight blob into LDS ----
#if HAVE_TDM
  if (wid == 0) {
    const unsigned long long ga = (unsigned long long)wall;
    // Generic LDS pointer low 32 bits == LDS byte offset (ISA 10.2 aperture map)
    const unsigned ldsoff = (unsigned)(unsigned long long)(const void*)Wall;
    u32x4 g0;
    g0[0] = 1u;                                          // count=1, user D#
    g0[1] = ldsoff;                                      // lds_addr
    g0[2] = (unsigned)(ga & 0xFFFFFFFFull);              // global_addr[31:0]
    g0[3] = (unsigned)((ga >> 32) & 0x1FFFFFFull)        // global_addr[56:32]
          | (2u << 30);                                  // type=2 ("image")
    i32x8 g1 = {};
    g1[0] = (int)(1u << 16);                             // data_size=1 (2 bytes)
    g1[1] = (int)(((unsigned)WTOT & 0xFFFFu) << 16);     // tensor_dim0[15:0]
    g1[2] = (int)(((unsigned)WTOT >> 16) | (1u << 16));  // dim0[31:16] | tensor_dim1=1
    g1[3] = (int)(((unsigned)WTOT & 0xFFFFu) << 16);     // tile_dim0 = WTOT
    g1[4] = 1;                                           // tile_dim1=1, tile_dim2=0
    g1[5] = (int)WTOT;                                   // tensor_dim0_stride
    i32x4 g2 = {};
    i32x4 g3 = {};
#if defined(__clang_major__) && (__clang_major__ >= 23)
    i32x8 g4 = {};
    __builtin_amdgcn_tensor_load_to_lds(g0, g1, g2, g3, g4, 0);
#else
    __builtin_amdgcn_tensor_load_to_lds(g0, g1, g2, g3, 0);
#endif
  }
#else
  for (int i = tid; i < WTOT; i += 256) Wall[i] = wall[i];
#endif

  // ---- ball query: append within-radius candidates to LDS buffer ----
  for (int p = tid; p < Nn; p += 256) {
    const size_t o = ((size_t)b * Nn + p) * 3;
    const float dx = pos[o] - pcx, dy = pos[o + 1] - pcy, dz = pos[o + 2] - pcz;
    const float d2 = dx * dx + dy * dy + dz * dz;
    if (d2 <= R2c) {
      const int slot = atomicAdd(&candCount, 1);
      if (slot < CAND_MAX) { candD[slot] = d2; candI[slot] = p; }
    }
  }
  if (tid < 64)  { b1f[tid] = b1[tid]; b2f[tid] = b2[tid]; }
  if (tid < 128) b3f[tid] = b3[tid];
  __syncthreads();

  // ---- wave 0: select K smallest-d2 candidates (== topK ∩ radius) ----
  if (wid == 0) {
    int cc = candCount; if (cc > CAND_MAX) cc = CAND_MAX;
    int sel = cc < Kk ? cc : Kk;
    if (lane == 0) sCnt = sel;
    for (int r = 0; r < sel; ++r) {
      float bd = INFINITY; int bp = 0; int bg = 0x7fffffff;
      for (int i = lane; i < cc; i += 32) {
        const float d = candD[i];
        const int   g = candI[i];
        if (d < bd || (d == bd && g < bg)) { bd = d; bp = i; bg = g; }
      }
#pragma unroll
      for (int o = 16; o > 0; o >>= 1) {
        const float od = __shfl_xor(bd, o);
        const int   op = __shfl_xor(bp, o);
        const int   og = __shfl_xor(bg, o);
        if (od < bd || (od == bd && og < bg)) { bd = od; bp = op; bg = og; }
      }
      if (lane == 0) { nbrS[r] = bg; candD[bp] = INFINITY; }
      __threadfence_block();
    }
#if HAVE_TDM
    __builtin_amdgcn_s_wait_tensorcnt(0);   // weights resident before barrier
#endif
  }
  __syncthreads();

  const int cnt = sCnt;

  // ---- prefetch selected neighbor feature rows ----
  if (tid < Kk && tid < cnt)
    __builtin_prefetch(&x[((size_t)b * Nn + nbrS[tid]) * Ff], 0, 0);

  // ---- build H = [x_j | pos_j - pos_i | 0-pad] as f16, rows >= cnt zeroed ----
  for (int i = tid; i < 64 * 96; i += 256) {
    const int r = i / 96, c = i % 96;
    float v = 0.0f;
    if (r < cnt && c < Ff + 3) {
      const int    gp = nbrS[r];
      const size_t gb = (size_t)b * Nn + gp;
      if (c < Ff) v = x[gb * Ff + c];
      else {
        const int d = c - Ff;
        const float pc = (d == 0) ? pcx : (d == 1) ? pcy : pcz;
        v = pos[gb * 3 + d] - pc;
      }
    }
    Hh[r * 96 + c] = (_Float16)v;
  }
  __syncthreads();

  // ---- MLP layer 1: [64x96] @ [96x64]; 16 tiles, 2 per wave, straight-line ----
#pragma unroll
  for (int it = 0; it < 2; ++it) {
    const int t  = wid + it * 8;
    const int mt = t >> 2, nt = t & 3;
    v8f acc = {};
    mlp_tile(Hh + mt * 16 * 96, 96, W1t + (nt * 16) * 96, 96, 3, acc);
    store_c_bias_relu_f16(O1h + (mt * 16) * 64 + nt * 16, 64, acc, &b1f[nt * 16]);
  }
  __syncthreads();

  // ---- MLP layer 2: [64x64] @ [64x64] ----
#pragma unroll
  for (int it = 0; it < 2; ++it) {
    const int t  = wid + it * 8;
    const int mt = t >> 2, nt = t & 3;
    v8f acc = {};
    mlp_tile(O1h + mt * 16 * 64, 64, W2t + (nt * 16) * 64, 64, 2, acc);
    store_c_bias_relu_f16(O2h + (mt * 16) * 64 + nt * 16, 64, acc, &b2f[nt * 16]);
  }
  __syncthreads();

  // ---- MLP layer 3: [64x64] @ [64x128] + masked relu-max over neighbors ----
#pragma unroll
  for (int j = 0; j < 4; ++j) {
    const int t  = wid * 4 + j;
    const int mt = t >> 3, nt = t & 7;
    v8f acc = {};
    mlp_tile(O2h + mt * 16 * 64, 64, W3t + (nt * 16) * 64, 64, 2, acc);
    const int half = lane >> 4, n = lane & 15;
    float m = -INFINITY;
#pragma unroll
    for (int r = 0; r < 8; ++r) {
      const int row = mt * 16 + r + 8 * half;
      if (row < cnt) m = fmaxf(m, fmaxf(acc[r] + b3f[nt * 16 + n], 0.0f));
    }
    m = fmaxf(m, __shfl_xor(m, 16));
    if (lane < 16) pmax[mt * 128 + nt * 16 + lane] = m;
  }
  __syncthreads();

  if (tid < H3c) {
    const float v = fmaxf(fmaxf(pmax[0 * 128 + tid], pmax[1 * 128 + tid]),
                          fmaxf(pmax[2 * 128 + tid], pmax[3 * 128 + tid]));
    xOut[(size_t)cid * H3c + tid] = v;
  }
}

// ---------------------------------------------------------------------------
extern "C" void kernel_launch(void* const* d_in, const int* in_sizes, int n_in,
                              void* d_out, int out_size, void* d_ws, size_t ws_size,
                              hipStream_t stream) {
  (void)in_sizes; (void)n_in; (void)out_size; (void)ws_size;
  const float* x   = (const float*)d_in[0];
  const float* pos = (const float*)d_in[1];
  // d_in[2] = batch (int32), unused: batch_out == cloud id for equal-size clouds
  const float* W1 = (const float*)d_in[3];
  const float* b1 = (const float*)d_in[4];
  const float* W2 = (const float*)d_in[5];
  const float* b2 = (const float*)d_in[6];
  const float* W3 = (const float*)d_in[7];
  const float* b3 = (const float*)d_in[8];

  float* out      = (float*)d_out;
  float* xOut     = out;                                  // [B*S, 128]
  float* posOut   = out + (size_t)Bc * Ss * H3c;          // [B*S, 3]
  float* batchOut = posOut + (size_t)Bc * Ss * 3;         // [B*S]

  _Float16* wpacked = (_Float16*)d_ws;                    // 36864 bytes used

  pack_weights_kernel<<<(WTOT + 255) / 256, 256, 0, stream>>>(W1, W2, W3, wpacked);
  fps_kernel<<<Bc, 1024, 0, stream>>>(pos, posOut, batchOut);
  sa_mlp_kernel<<<Bc * Ss, 256, 0, stream>>>(x, pos, wpacked, b1, b2, b3,
                                             posOut, xOut);
}
